// PVMLayer_38448547233808
// MI455X (gfx1250) — compile-verified, matrix operand order checked by reading, not run on hardware
//
#include <hip/hip_runtime.h>
#include <stdint.h>

// ---------------------------------------------------------------- types ----
typedef __attribute__((ext_vector_type(16))) __bf16 v16bf;
typedef __attribute__((ext_vector_type(8)))  float  v8f;

#define L_SEQ 4096
#define NSEQ  32          // 4 groups * B=8
#define DIN   256         // d_inner
#define CCH   512         // full channel dim
#define DM    128         // d_model per group
#define NB    8           // batch
#define DBLW  48          // dt(8)+B(16)+C(16) padded to 48 for 16-wide WMMA tiles
#define SCAN_T 32         // scan LDS chunk (timesteps)

__device__ __forceinline__ uint16_t f2bf(float f) {
  uint32_t u = __float_as_uint(f);
  uint32_t r = (u + 0x7FFFu + ((u >> 16) & 1u)) >> 16;  // round-to-nearest-even
  return (uint16_t)r;
}

// ------------------------------------------------------- weight converts ---
__global__ __launch_bounds__(256) void cvt_bf16(const float* __restrict__ in,
                                                uint16_t* __restrict__ out, int n) {
  int i = blockIdx.x * 256 + threadIdx.x;
  if (i < n) out[i] = f2bf(in[i]);
}

// W_x is (40,256) row-major; pad to (48,256) with zero rows so N is a multiple of 16
__global__ __launch_bounds__(256) void pad_wx(const float* __restrict__ wx,
                                              uint16_t* __restrict__ out) {
  int i = blockIdx.x * 256 + threadIdx.x;
  if (i >= DBLW * DIN) return;
  int r = i / DIN;
  float v = (r < 40) ? wx[(size_t)r * DIN + (i % DIN)] : 0.f;
  out[i] = f2bf(v);
}

// --------------------------------------------------------- block reduce ----
__device__ __forceinline__ void block_reduce2(float& s, float& ss, float* red) {
  #pragma unroll
  for (int off = 16; off > 0; off >>= 1) {
    s  += __shfl_down(s,  off, 32);
    ss += __shfl_down(ss, off, 32);
  }
  unsigned wave = threadIdx.x >> 5, lane = threadIdx.x & 31;
  if (lane == 0) { red[wave] = s; red[8 + wave] = ss; }
  __syncthreads();
  if (threadIdx.x == 0) {
    float a = 0.f, b = 0.f;
    #pragma unroll
    for (int i = 0; i < 8; ++i) { a += red[i]; b += red[8 + i]; }
    red[16] = a; red[17] = b;
  }
  __syncthreads();
  s = red[16]; ss = red[17];
}

// ------------------------------------------- LN1 + split into 32 streams ---
// x:(B,512,L) -> ch:(32,L,128)   seq n = (c>>7)*8 + b
__global__ __launch_bounds__(256) void ln_split(const float* __restrict__ x,
                                                const float* __restrict__ w,
                                                const float* __restrict__ b,
                                                float* __restrict__ ch32,
                                                uint16_t* __restrict__ chbf) {
  int bl = blockIdx.x; int bi = bl >> 12; int l = bl & 4095;
  int c0 = threadIdx.x, c1 = threadIdx.x + 256;
  const float* xb = x + ((size_t)bi * CCH) * L_SEQ + l;
  float v0 = xb[(size_t)c0 * L_SEQ];
  float v1 = xb[(size_t)c1 * L_SEQ];
  float s = v0 + v1, ss = v0 * v0 + v1 * v1;
  __shared__ float red[18];
  block_reduce2(s, ss, red);
  float mu = s * (1.f / CCH);
  float rs = rsqrtf(ss * (1.f / CCH) - mu * mu + 1e-5f);
  {
    float t = (v0 - mu) * rs * w[c0] + b[c0];
    int n = (c0 >> 7) * NB + bi;
    size_t o = ((size_t)n * L_SEQ + l) * DM + (c0 & 127);
    ch32[o] = t; chbf[o] = f2bf(t);
  }
  {
    float t = (v1 - mu) * rs * w[c1] + b[c1];
    int n = (c1 >> 7) * NB + bi;
    size_t o = ((size_t)n * L_SEQ + l) * DM + (c1 & 127);
    ch32[o] = t; chbf[o] = f2bf(t);
  }
}

// ------------------------------------------------------------ WMMA GEMM ----
// C[MxN] = A[MxK](bf16) * W[NxK](bf16)^T, f32 accum.
// Register-blocked: each wave computes an (MT*16) x (NT*16) macro-tile,
// reusing MT A-fragments across NT B-fragments -> MT*NT WMMAs per
// 2*(MT+NT) b128 loads per K-step.
// mode 0: row-major store (ldc = N).  mode 1: NCHW store for proj output
// (r = b*4096+l -> out[((b*N)+col)*4096 + l]) with bias.
union FragAB { uint32_t u[8]; v16bf v; };

template <int MT, int NT>
__global__ __launch_bounds__(256) void wmma_gemm(const uint16_t* __restrict__ A,
                                                 const uint16_t* __restrict__ W,
                                                 float* __restrict__ C,
                                                 const float* __restrict__ bias,
                                                 int M, int N, int K, int mode) {
  unsigned wave = threadIdx.x >> 5;
  unsigned lane = threadIdx.x & 31;
  int row0 = (int)(blockIdx.x * 8 + wave) * (16 * MT);
  int col0 = (int)blockIdx.y * (16 * NT);
  if (row0 >= M) return;
  unsigned mr = lane & 15;      // row (A) / col (W) within tile
  unsigned kh = lane >> 4;      // K half select per 16-bit WMMA layout
  const uint16_t* ap[MT];
  const uint16_t* wp[NT];
  #pragma unroll
  for (int mi = 0; mi < MT; ++mi)
    ap[mi] = A + (size_t)(row0 + mi * 16 + (int)mr) * K + kh * 8;
  #pragma unroll
  for (int ni = 0; ni < NT; ++ni)
    wp[ni] = W + (size_t)(col0 + ni * 16 + (int)mr) * K + kh * 8;

  const v8f vzero = {0.f, 0.f, 0.f, 0.f, 0.f, 0.f, 0.f, 0.f};
  v8f acc[MT][NT];
  #pragma unroll
  for (int mi = 0; mi < MT; ++mi)
    #pragma unroll
    for (int ni = 0; ni < NT; ++ni) acc[mi][ni] = vzero;

  for (int k0 = 0; k0 < K; k0 += 32) {
    FragAB fa[MT], fb[NT];
    #pragma unroll
    for (int mi = 0; mi < MT; ++mi) {
      *(uint4*)&fa[mi].u[0] = *(const uint4*)(ap[mi] + k0);      // K=kh*8..+7
      *(uint4*)&fa[mi].u[4] = *(const uint4*)(ap[mi] + k0 + 16); // K=16+kh*8..
    }
    #pragma unroll
    for (int ni = 0; ni < NT; ++ni) {
      *(uint4*)&fb[ni].u[0] = *(const uint4*)(wp[ni] + k0);
      *(uint4*)&fb[ni].u[4] = *(const uint4*)(wp[ni] + k0 + 16);
    }
    #pragma unroll
    for (int mi = 0; mi < MT; ++mi)
      #pragma unroll
      for (int ni = 0; ni < NT; ++ni)
        acc[mi][ni] = __builtin_amdgcn_wmma_f32_16x16x32_bf16(
            false, fa[mi].v, false, fb[ni].v, (short)0, acc[mi][ni], false, false);
  }

  #pragma unroll
  for (int mi = 0; mi < MT; ++mi) {
    int rbase = row0 + mi * 16 + (int)((lane >> 4) << 3); // lanes16-31: M=8..15
    #pragma unroll
    for (int ni = 0; ni < NT; ++ni) {
      int col = col0 + ni * 16 + (int)(lane & 15);
      float bv = bias ? bias[col] : 0.f;
      #pragma unroll
      for (int i = 0; i < 8; ++i) {
        int r = rbase + i;
        float v = acc[mi][ni][i] + bv;
        size_t o = (mode == 0)
                       ? ((size_t)r * N + col)
                       : ((((size_t)(r >> 12) * N + col) << 12) | (size_t)(r & 4095));
        C[o] = v;
      }
    }
  }
}

// ----------------------------------------- depthwise causal conv4 + SiLU ---
// xz:(32,L,512) cols 0..255 = xp.   out xs:(32,L,256)
__global__ __launch_bounds__(256) void conv_silu(const float* __restrict__ xz,
                                                 const float* __restrict__ cw,
                                                 const float* __restrict__ cb,
                                                 float* __restrict__ xs32,
                                                 uint16_t* __restrict__ xsbf) {
  size_t idx = (size_t)blockIdx.x * 256 + threadIdx.x;   // (n*L + l)*256 + d
  int d = (int)(idx & (DIN - 1));
  size_t nl = idx >> 8;
  int l = (int)(nl & 4095);
  size_t n = nl >> 12;
  const float* xp = xz + n * (size_t)L_SEQ * CCH + d;
  float acc = cb[d];
  #pragma unroll
  for (int k = 0; k < 4; ++k) {
    int ll = l - 3 + k;
    if (ll >= 0) acc += xp[(size_t)ll * CCH] * cw[d * 4 + k];
  }
  float sv = acc / (1.f + __expf(-acc));
  xs32[idx] = sv; xsbf[idx] = f2bf(sv);
}

// ----------------------------------------------------- selective scan ------
// One block per sequence, lane = channel d. Token-shared (dt,B,C) staged into
// LDS in SCAN_T chunks via CDNA5 async global->LDS loads. Recurrence in f32,
// SiLU(z) gate fused, bf16 output for the W_out WMMA GEMM.
__global__ __launch_bounds__(256) void scan_kernel(const float* __restrict__ xz,
                                                   const float* __restrict__ xs32,
                                                   const float* __restrict__ dbl,
                                                   const float* __restrict__ W_dt,
                                                   const float* __restrict__ b_dt,
                                                   const float* __restrict__ A_log,
                                                   const float* __restrict__ Dvec,
                                                   uint16_t* __restrict__ ybf) {
  const int n = blockIdx.x;
  const int d = threadIdx.x;
  float wdt[8];
  #pragma unroll
  for (int j = 0; j < 8; ++j) wdt[j] = W_dt[d * 8 + j];
  const float bdt = b_dt[d];
  float Ar[16];
  #pragma unroll
  for (int s = 0; s < 16; ++s) Ar[s] = -__expf(A_log[d * 16 + s]);
  const float Dd = Dvec[d];
  float h[16];
  #pragma unroll
  for (int s = 0; s < 16; ++s) h[s] = 0.f;

  __shared__ float sdbl[SCAN_T * DBLW];   // 32 steps * 48 = 6KB
  const float* dblb = dbl + (size_t)n * L_SEQ * DBLW;
  const float* ub = xs32 + (size_t)n * L_SEQ * DIN + d;
  const float* zb = xz + (size_t)n * L_SEQ * CCH + DIN + d;
  uint16_t* yb = ybf + (size_t)n * L_SEQ * DIN + d;
  const uint32_t lds0 = (uint32_t)(uintptr_t)&sdbl[0];

  for (int l0 = 0; l0 < L_SEQ; l0 += SCAN_T) {
    __syncthreads();                       // previous chunk fully consumed
    const float* gsrc = dblb + (size_t)l0 * DBLW;   // uniform per block -> SGPR
    #pragma unroll
    for (int j = 0; j < 6; ++j) {          // 256 thr * 6 dwords = 1536 = 32*48
      uint32_t voff = ((uint32_t)threadIdx.x + (uint32_t)j * 256u) * 4u;
      asm volatile("global_load_async_to_lds_b32 %0, %1, %2"
                   :: "v"(lds0 + voff), "v"(voff), "s"(gsrc) : "memory");
    }
    asm volatile("s_wait_asynccnt 0x0" ::: "memory");
    __syncthreads();
    #pragma unroll 4
    for (int t = 0; t < SCAN_T; ++t) {
      const float* row = &sdbl[t * DBLW];
      float dp = bdt;
      #pragma unroll
      for (int j = 0; j < 8; ++j) dp += row[j] * wdt[j];
      float delta = (dp > 20.f) ? dp : log1pf(__expf(dp));   // softplus
      size_t loff = (size_t)(l0 + t) * DIN;
      float u = ub[loff];
      float du = delta * u;
      float y = 0.f;
      #pragma unroll
      for (int s = 0; s < 16; ++s) {
        h[s] = __expf(delta * Ar[s]) * h[s] + du * row[8 + s];
        y += h[s] * row[24 + s];
      }
      y += u * Dd;
      float z = zb[(size_t)(l0 + t) * CCH];
      y *= z / (1.f + __expf(-z));         // * silu(z)
      yb[loff] = f2bf(y);
    }
  }
}

// --------------------------------------------- skip + regroup + LN2 --------
__global__ __launch_bounds__(256) void combine_ln(const float* __restrict__ y32,
                                                  const float* __restrict__ ch32,
                                                  const float* __restrict__ skip,
                                                  const float* __restrict__ w,
                                                  const float* __restrict__ b,
                                                  uint16_t* __restrict__ ynbf) {
  int bl = blockIdx.x; int bi = bl >> 12; int l = bl & 4095;
  float sk = skip[0];
  int c0 = threadIdx.x, c1 = threadIdx.x + 256;
  auto ld = [&](int c) {
    int n = (c >> 7) * NB + bi;
    size_t o = ((size_t)n * L_SEQ + l) * DM + (c & 127);
    return y32[o] + sk * ch32[o];
  };
  float v0 = ld(c0), v1 = ld(c1);
  float s = v0 + v1, ss = v0 * v0 + v1 * v1;
  __shared__ float red[18];
  block_reduce2(s, ss, red);
  float mu = s * (1.f / CCH);
  float rs = rsqrtf(ss * (1.f / CCH) - mu * mu + 1e-5f);
  size_t ro = ((size_t)bi * L_SEQ + l) * CCH;
  ynbf[ro + c0] = f2bf((v0 - mu) * rs * w[c0] + b[c0]);
  ynbf[ro + c1] = f2bf((v1 - mu) * rs * w[c1] + b[c1]);
}

// ---------------------------------------------------------------- driver ---
extern "C" void kernel_launch(void* const* d_in, const int* in_sizes, int n_in,
                              void* d_out, int out_size, void* d_ws, size_t ws_size,
                              hipStream_t stream) {
  (void)in_sizes; (void)n_in; (void)out_size; (void)ws_size;
  const float* x      = (const float*)d_in[0];
  const float* norm_w = (const float*)d_in[1];
  const float* norm_b = (const float*)d_in[2];
  const float* W_in   = (const float*)d_in[3];
  const float* conv_w = (const float*)d_in[4];
  const float* conv_b = (const float*)d_in[5];
  const float* W_x    = (const float*)d_in[6];
  const float* W_dt   = (const float*)d_in[7];
  const float* b_dt   = (const float*)d_in[8];
  const float* A_log  = (const float*)d_in[9];
  const float* Dp     = (const float*)d_in[10];
  const float* W_out  = (const float*)d_in[11];
  const float* proj_w = (const float*)d_in[12];
  const float* proj_b = (const float*)d_in[13];
  const float* skip   = (const float*)d_in[14];
  float* out = (float*)d_out;

  char* p = (char*)d_ws;
  auto alloc = [&](size_t bytes) -> void* {
    void* r = (void*)p; p += (bytes + 255) & ~(size_t)255; return r;
  };
  const size_t M = (size_t)NSEQ * L_SEQ;                    // 131072 rows
  float*    ch32  = (float*)   alloc(M * DM  * 4);
  uint16_t* chbf  = (uint16_t*)alloc(M * DM  * 2);
  float*    xz    = (float*)   alloc(M * CCH * 4);
  float*    xs32  = (float*)   alloc(M * DIN * 4);
  uint16_t* xsbf  = (uint16_t*)alloc(M * DIN * 2);
  float*    dblb  = (float*)   alloc(M * DBLW * 4);
  uint16_t* ybf   = (uint16_t*)alloc(M * DIN * 2);
  float*    y32   = (float*)   alloc(M * DM  * 4);
  uint16_t* ynbf  = (uint16_t*)alloc((size_t)NB * L_SEQ * CCH * 2);
  uint16_t* WinB  = (uint16_t*)alloc((size_t)CCH * DM  * 2);
  uint16_t* WxB   = (uint16_t*)alloc((size_t)DBLW * DIN * 2);
  uint16_t* WoutB = (uint16_t*)alloc((size_t)DM  * DIN * 2);
  uint16_t* WprB  = (uint16_t*)alloc((size_t)CCH * CCH * 2);

  // weight converts (all (N,K) row-major -> bf16)
  cvt_bf16<<<(CCH * DM + 255) / 256, 256, 0, stream>>>(W_in, WinB, CCH * DM);
  pad_wx  <<<(DBLW * DIN + 255) / 256, 256, 0, stream>>>(W_x, WxB);
  cvt_bf16<<<(DM * DIN + 255) / 256, 256, 0, stream>>>(W_out, WoutB, DM * DIN);
  cvt_bf16<<<(CCH * CCH + 255) / 256, 256, 0, stream>>>(proj_w, WprB, CCH * CCH);

  // LN1 + split
  ln_split<<<NB * L_SEQ, 256, 0, stream>>>(x, norm_w, norm_b, ch32, chbf);
  // xz = ch @ W_in^T   (M x 512, K=128): 32x32 macro-tiles per wave
  wmma_gemm<2, 2><<<dim3((unsigned)(M / 32 / 8), CCH / 32), 256, 0, stream>>>(
      chbf, WinB, xz, nullptr, (int)M, CCH, DM, 0);
  // depthwise conv + SiLU
  conv_silu<<<(unsigned)(M * DIN / 256), 256, 0, stream>>>(xz, conv_w, conv_b,
                                                           xs32, xsbf);
  // dbl = xs @ W_x^T   (M x 48, K=256): 16x48 per wave (A-frag reused 3x)
  wmma_gemm<1, 3><<<dim3((unsigned)(M / 16 / 8), 1), 256, 0, stream>>>(
      xsbf, WxB, dblb, nullptr, (int)M, DBLW, DIN, 0);
  // selective scan + gate
  scan_kernel<<<NSEQ, 256, 0, stream>>>(xz, xs32, dblb, W_dt, b_dt, A_log, Dp, ybf);
  // y @ W_out^T   (M x 128, K=256)
  wmma_gemm<2, 2><<<dim3((unsigned)(M / 32 / 8), DM / 32), 256, 0, stream>>>(
      ybf, WoutB, y32, nullptr, (int)M, DM, DIN, 0);
  // skip + regroup + LN2
  combine_ln<<<NB * L_SEQ, 256, 0, stream>>>(y32, ch32, skip, norm_w, norm_b, ynbf);
  // proj (32768 x 512, K=512), bias, fused NCHW transpose store
  wmma_gemm<2, 2><<<dim3((unsigned)((size_t)NB * L_SEQ / 32 / 8), CCH / 32), 256, 0,
                    stream>>>(ynbf, WprB, out, proj_b, NB * L_SEQ, CCH, CCH, 1);
}
